// BahdanauAttention_49795850830321
// MI455X (gfx1250) — compile-verified, matrix operand order checked by reading, not run on hardware
//
#include <hip/hip_runtime.h>
#include <hip/hip_bf16.h>
#include <math.h>

#define B_DIM 64
#define S_DIM 2048
#define H_DIM 512
#define NEG_INF -1e9f

#define M_TILE 64                        // token rows per workgroup (4 x 16)
#define NCHUNK 8                         // S-chunks for context pass
#define CHUNK  (S_DIM / NCHUNK)          // 256

typedef __attribute__((ext_vector_type(16))) __bf16 v16bf;
typedef __attribute__((ext_vector_type(8)))  float  v8f;
typedef __attribute__((ext_vector_type(4)))  unsigned int u32x4;

union BfFrag {
    v16bf bf;
    u32x4 u[2];
};

__device__ __forceinline__ unsigned short f32_to_bf16_rne(float x) {
    unsigned int u = __float_as_uint(x);
    unsigned int r = u + 0x7FFFu + ((u >> 16) & 1u);
    return (unsigned short)(r >> 16);
}

// tanh(x) = 1 - 2/(exp(2x)+1): 2 transcendentals (v_exp_f32, v_rcp_f32),
// correct limits (+/-1), vastly cheaper than libm tanhf.
__device__ __forceinline__ float fast_tanh(float x) {
    float t = __expf(2.0f * x);
    float r = __builtin_amdgcn_rcpf(t + 1.0f);
    return fmaf(-2.0f, r, 1.0f);
}

// ---------------------------------------------------------------------------
// Kernel 0: W_attn fp32 -> bf16 (512 KB, stays resident in L2 for the GEMM)
// ---------------------------------------------------------------------------
__global__ __launch_bounds__(256)
void convert_w_kernel(const float* __restrict__ W, unsigned short* __restrict__ Wbf) {
    int i = blockIdx.x * blockDim.x + threadIdx.x;
    if (i < H_DIM * H_DIM) Wbf[i] = f32_to_bf16_rne(W[i]);
}

// ---------------------------------------------------------------------------
// Kernel 1: scores[b,s] = v_w . tanh(hidden[b,s,:] @ W^T + b_attn)
// One workgroup (8 waves) per 64 token rows. Each wave owns a 4(M) x 4(N)
// grid of 16x16 WMMA tiles: 16 WMMAs per K-step vs 16 x b128 loads.
// ---------------------------------------------------------------------------
__global__ __launch_bounds__(256)
void scores_kernel(const float* __restrict__ hidden,
                   const unsigned short* __restrict__ Wbf,
                   const float* __restrict__ b_attn,
                   const float* __restrict__ v_w,
                   float* __restrict__ scores) {
    constexpr int LDS_STRIDE = H_DIM + 8;                   // pad to dodge bank conflicts
    __shared__ alignas(16) unsigned short sA[M_TILE * LDS_STRIDE];   // ~66.5 KB
    __shared__ float sScore[M_TILE];

    const int tid = threadIdx.x;
    const size_t rowBlock = blockIdx.x;                     // 64 token rows per block
    const float* gA = hidden + rowBlock * (size_t)(M_TILE * H_DIM);

    if (tid < M_TILE) sScore[tid] = 0.0f;

    // Stage 64x512 fp32 tile -> bf16 LDS (float4 loads, each thread 32x float4)
    {
        const float4* gA4 = (const float4*)gA;
        #pragma unroll
        for (int i = 0; i < (M_TILE * H_DIM / 4) / 256; ++i) {
            int f4 = i * 256 + tid;
            float4 v = gA4[f4];
            int e = f4 * 4;
            int r = e >> 9;                                 // / 512
            int c = e & (H_DIM - 1);
            unsigned short* dst = &sA[r * LDS_STRIDE + c];
            dst[0] = f32_to_bf16_rne(v.x);
            dst[1] = f32_to_bf16_rne(v.y);
            dst[2] = f32_to_bf16_rne(v.z);
            dst[3] = f32_to_bf16_rne(v.w);
        }
    }
    __syncthreads();

    const int wave = tid >> 5;
    const int lane = tid & 31;
    const int half = lane >> 4;      // K sub-offset selector per ISA A-frag layout
    const int ln   = lane & 15;      // A: M row within tile; B: N column within tile

    v8f acc[16] = {};                // acc[mt*4 + i]: M-tile mt, N-tile (wave*4 + i)

    for (int kk = 0; kk < H_DIM; kk += 32) {
        const int kb = kk + half * 8;

        // 4 A-fragments (one per M-tile) from LDS
        BfFrag a[4];
        #pragma unroll
        for (int mt = 0; mt < 4; ++mt) {
            const unsigned short* ap = &sA[(mt * 16 + ln) * LDS_STRIDE + kb];
            a[mt].u[0] = *(const u32x4*)(ap);
            a[mt].u[1] = *(const u32x4*)(ap + 16);
        }

        // For each N-tile: load B-fragment once, reuse across 4 M-tiles
        #pragma unroll
        for (int i = 0; i < 4; ++i) {
            const int nrow = (wave * 4 + i) * 16 + ln;      // row of W == output col n
            const unsigned short* wp = Wbf + (size_t)nrow * H_DIM + kb;
            BfFrag b;
            b.u[0] = *(const u32x4*)(wp);
            b.u[1] = *(const u32x4*)(wp + 16);
            #pragma unroll
            for (int mt = 0; mt < 4; ++mt) {
                acc[mt * 4 + i] = __builtin_amdgcn_wmma_f32_16x16x32_bf16(
                    false, a[mt].bf, false, b.bf, (short)0, acc[mt * 4 + i],
                    false, false);
            }
        }
    }

    // Epilogue: part[m] += tanh(D[m,n] + b_attn[n]) * v_w[n]
    // C/D layout: VGPR j -> M = half*8 + j (within tile); lane&15 -> N within tile
    #pragma unroll
    for (int mt = 0; mt < 4; ++mt) {
        float part[8] = {0.f, 0.f, 0.f, 0.f, 0.f, 0.f, 0.f, 0.f};
        #pragma unroll
        for (int i = 0; i < 4; ++i) {
            const int ng = (wave * 4 + i) * 16 + ln;
            const float vw = v_w[ng];
            const float bb = b_attn[ng];
            #pragma unroll
            for (int j = 0; j < 8; ++j)
                part[j] += fast_tanh(acc[mt * 4 + i][j] + bb) * vw;
        }
        #pragma unroll
        for (int j = 0; j < 8; ++j)
            atomicAdd(&sScore[mt * 16 + half * 8 + j], part[j]);    // ds_add_f32
    }
    __syncthreads();

    if (tid < M_TILE)
        scores[rowBlock * M_TILE + tid] = sScore[tid];
}

// ---------------------------------------------------------------------------
// Kernel 2a: masked softmax over S per batch -> weights (written to d_out)
// ---------------------------------------------------------------------------
__global__ __launch_bounds__(512)
void softmax_kernel(const unsigned char* __restrict__ mask,
                    const float* __restrict__ scores,
                    float* __restrict__ w_out) {
    __shared__ float sw[S_DIM];
    __shared__ float red[16];
    const int tid  = threadIdx.x;
    const int b    = blockIdx.x;
    const int lane = tid & 31;
    const int wave = tid >> 5;

    for (int s = tid; s < S_DIM; s += 512) {
        float sc = scores[(size_t)b * S_DIM + s];
        sw[s] = mask[(size_t)b * S_DIM + s] ? sc : NEG_INF;
    }
    __syncthreads();

    // max reduction
    float mx = -INFINITY;
    #pragma unroll
    for (int i = 0; i < S_DIM / 512; ++i) mx = fmaxf(mx, sw[tid + i * 512]);
    #pragma unroll
    for (int off = 16; off > 0; off >>= 1) mx = fmaxf(mx, __shfl_xor(mx, off, 32));
    if (lane == 0) red[wave] = mx;
    __syncthreads();
    if (tid == 0) {
        float v = red[0];
        for (int w = 1; w < 16; ++w) v = fmaxf(v, red[w]);
        red[0] = v;
    }
    __syncthreads();
    mx = red[0];
    __syncthreads();                                        // before red is reused

    // exp + sum reduction
    float lsum = 0.0f;
    #pragma unroll
    for (int i = 0; i < S_DIM / 512; ++i) {
        int s = tid + i * 512;
        float e = __expf(sw[s] - mx);
        sw[s] = e;
        lsum += e;
    }
    #pragma unroll
    for (int off = 16; off > 0; off >>= 1) lsum += __shfl_xor(lsum, off, 32);
    if (lane == 0) red[wave] = lsum;
    __syncthreads();
    if (tid == 0) {
        float t = 0.0f;
        for (int w = 0; w < 16; ++w) t += red[w];
        red[0] = 1.0f / t;
    }
    __syncthreads();
    const float inv = red[0];

    #pragma unroll
    for (int i = 0; i < S_DIM / 512; ++i) {
        int s = tid + i * 512;
        w_out[(size_t)b * S_DIM + s] = sw[s] * inv;
    }
}

// ---------------------------------------------------------------------------
// Kernel 2b: partial context over an S-chunk. Grid (NCHUNK, B) = 512 blocks
// keeps the whole chip busy on the 256 MB streaming read. Deterministic
// (no float atomics): partials land in workspace, reduced by kernel 2c.
// ---------------------------------------------------------------------------
__global__ __launch_bounds__(512)
void context_partial_kernel(const float* __restrict__ hidden,
                            const float* __restrict__ w_out,
                            float* __restrict__ ctx_part) {
    __shared__ float swc[CHUNK];
    const int tid = threadIdx.x;
    const int c   = blockIdx.x;          // S-chunk
    const int b   = blockIdx.y;          // batch

    if (tid < CHUNK)
        swc[tid] = w_out[(size_t)b * S_DIM + c * CHUNK + tid];
    __syncthreads();

    float acc = 0.0f;
    const float* hb = hidden + ((size_t)b * S_DIM + (size_t)c * CHUNK) * H_DIM + tid;
    #pragma unroll 4
    for (int s = 0; s < CHUNK; ++s)
        acc = fmaf(swc[s], hb[(size_t)s * H_DIM], acc);

    ctx_part[((size_t)b * NCHUNK + c) * H_DIM + tid] = acc;
}

// ---------------------------------------------------------------------------
// Kernel 2c: reduce NCHUNK partials -> context [B, H]
// ---------------------------------------------------------------------------
__global__ __launch_bounds__(256)
void context_reduce_kernel(const float* __restrict__ ctx_part,
                           float* __restrict__ ctx_out) {
    int idx = blockIdx.x * blockDim.x + threadIdx.x;        // [0, B*H)
    int b = idx >> 9;                                       // / H_DIM
    int h = idx & (H_DIM - 1);
    float s = 0.0f;
    #pragma unroll
    for (int c = 0; c < NCHUNK; ++c)
        s += ctx_part[((size_t)b * NCHUNK + c) * H_DIM + h];
    ctx_out[idx] = s;
}

// ---------------------------------------------------------------------------
extern "C" void kernel_launch(void* const* d_in, const int* in_sizes, int n_in,
                              void* d_out, int out_size, void* d_ws, size_t ws_size,
                              hipStream_t stream) {
    (void)in_sizes; (void)n_in; (void)out_size; (void)ws_size;

    const float*         hidden = (const float*)d_in[0];
    const unsigned char* mask   = (const unsigned char*)d_in[1];
    const float*         W_attn = (const float*)d_in[2];
    const float*         b_attn = (const float*)d_in[3];
    const float*         v_w    = (const float*)d_in[4];

    char* ws = (char*)d_ws;
    unsigned short* Wbf      = (unsigned short*)ws;                       // 512 KB
    float*          scores   = (float*)(ws + (size_t)H_DIM * H_DIM * 2);  // 512 KB
    float*          ctx_part = scores + (size_t)B_DIM * S_DIM;            // 1 MB

    float* ctx_out = (float*)d_out;                       // [B, H]
    float* w_out   = ctx_out + (size_t)B_DIM * H_DIM;     // [B, S]

    convert_w_kernel<<<(H_DIM * H_DIM) / 256, 256, 0, stream>>>(W_attn, Wbf);
    scores_kernel<<<(B_DIM * S_DIM) / M_TILE, 256, 0, stream>>>(hidden, Wbf, b_attn, v_w, scores);
    softmax_kernel<<<B_DIM, 512, 0, stream>>>(mask, scores, w_out);
    context_partial_kernel<<<dim3(NCHUNK, B_DIM), 512, 0, stream>>>(hidden, w_out, ctx_part);
    context_reduce_kernel<<<(B_DIM * H_DIM) / 256, 256, 0, stream>>>(ctx_part, ctx_out);
}
